// B_RNN_58686433132641
// MI455X (gfx1250) — compile-verified, matrix operand order checked by reading, not run on hardware
//
#include <hip/hip_runtime.h>
#include <hip/hip_bf16.h>
#include <stdint.h>

// ---------------------------------------------------------------------------
// Types for CDNA5 WMMA (wave32): bf16 A/B (v16bf = 8 VGPRs), f32 C/D (v8f)
// ---------------------------------------------------------------------------
typedef __attribute__((ext_vector_type(16))) __bf16 v16bf;
typedef __attribute__((ext_vector_type(8)))  float  v8f;

union Frag16 { v16bf v; uint32_t u[8]; };

static constexpr int Bsz = 8, Tlen = 4096, Dm = 256, Hm = 256;
static constexpr int VOCAB = 32000;
static constexpr int HROW = 258;               // padded LDS row stride (bf16 elems)
static constexpr size_t WT_ELEMS  = 4 * 256 * 256;         // 4 weight mats, bf16
static constexpr size_t WT_BYTES  = WT_ELEMS * 2;          // 512 KB
static constexpr size_t EMB_ELEMS = (size_t)VOCAB * Dm;    // 8.192M bf16
static constexpr size_t EMB_BYTES = EMB_ELEMS * 2;         // 16 MB

// K offset inside a 32-wide K-slab for the bf16 16x16x32 fragment layout:
// lanes 0-15: VGPR v<4 -> K=2v,2v+1 ; v>=4 -> K=16+2(v-4).. ; lanes 16-31: +8
__device__ __forceinline__ int kpat(int v, int hi) {
  return ((v < 4) ? (v * 2) : (16 + (v - 4) * 2)) + hi * 8;
}

__device__ __forceinline__ uint16_t f32_to_bf16_rne(float x) {
  uint32_t u = __float_as_uint(x);
  return (uint16_t)((u + 0x7fffu + ((u >> 16) & 1u)) >> 16);
}

__device__ __forceinline__ uint32_t pack2_bf16(float a, float b) {
  return (uint32_t)f32_to_bf16_rne(a) | ((uint32_t)f32_to_bf16_rne(b) << 16);
}

// ---------------------------------------------------------------------------
// Kernel 0a: transpose + convert the 4 weight matrices to bf16, layout [n][k]
// wt slot order: 0=W_hx, 1=W_hh, 2=W_hx_, 3=W_hh_
// ---------------------------------------------------------------------------
__global__ void __launch_bounds__(256)
rnn_wconv(const float* __restrict__ w0, const float* __restrict__ w1,
          const float* __restrict__ w2, const float* __restrict__ w3,
          uint16_t* __restrict__ wt) {
  int i = blockIdx.x * 256 + threadIdx.x;          // 0 .. 4*65536-1
  int w = i >> 16;
  int e = i & 65535;
  int n = e >> 8;
  int k = e & 255;
  const float* src = (w == 0) ? w0 : (w == 1) ? w1 : (w == 2) ? w2 : w3;
  wt[(size_t)w * 65536 + n * 256 + k] = f32_to_bf16_rne(src[k * 256 + n]);
}

// ---------------------------------------------------------------------------
// Kernel 0b: f32 -> bf16 copy of the embedding table (row-major, packed pairs)
// Memory-bound (48 MB traffic); keeps all conversion VALU out of the GEMM.
// ---------------------------------------------------------------------------
__global__ void __launch_bounds__(256)
rnn_econv(const float* __restrict__ emb, uint32_t* __restrict__ embb) {
  size_t p = (size_t)blockIdx.x * 256 + threadIdx.x;   // pair index
  if (p < EMB_ELEMS / 2) {
    float a = emb[p * 2], b = emb[p * 2 + 1];
    embb[p] = pack2_bf16(a, b);
  }
}

// ---------------------------------------------------------------------------
// Kernel 1: Z[dir,t,b,:] = emb[X[b,t]] @ W_hx(dir) + b_h(dir)
// One wave per 16x16 tile of the [32768 x 256] GEMM, per direction.
// A and B fragments are now pure b32 loads of pre-packed bf16 pairs.
// Output stored in C-fragment order:
//   zfrag[((dir*T + t)*16 + ntile)*16 + N][r] , r = b = 0..7  (8 floats/lane)
// ---------------------------------------------------------------------------
__global__ void __launch_bounds__(256)
rnn_zx_gemm(const int* __restrict__ X, const uint16_t* __restrict__ embb,
            const float* __restrict__ bh0, const float* __restrict__ bh1,
            const uint16_t* __restrict__ wt, float* __restrict__ zfrag) {
  const int wave = threadIdx.x >> 5;
  const int lane = threadIdx.x & 31;
  const int lo = lane & 15, hi = lane >> 4;

  const int tile  = blockIdx.x * 8 + wave;   // 0 .. 65535
  const int ntile = tile & 15;
  const int mtile = (tile >> 4) & 2047;
  const int dir   = tile >> 15;

  // A-side: this lane's row of the gathered (bf16) embedding
  const int row   = mtile * 16 + lo;         // m = t*8 + b
  const int t_row = row >> 3, b_row = row & 7;
  const int token = X[b_row * Tlen + t_row];
  const uint16_t* arow = embb + (size_t)token * Dm;

  // B-side: column j of W_hx(dir), bf16 [n][k] layout
  const int j = ntile * 16 + lo;
  const uint16_t* brow = wt + (dir ? 2 * 65536 : 0) + (size_t)j * 256;

  const float bias = (dir ? bh1 : bh0)[j];
  v8f acc;
#pragma unroll
  for (int r = 0; r < 8; ++r) acc[r] = bias;

#pragma unroll
  for (int kk = 0; kk < 8; ++kk) {
    Frag16 A, Bm;
#pragma unroll
    for (int v = 0; v < 8; ++v) {
      const int k = kk * 32 + kpat(v, hi);   // even -> 4B aligned
      A.u[v]  = *(const uint32_t*)(arow + k);
      Bm.u[v] = *(const uint32_t*)(brow + k);
    }
    acc = __builtin_amdgcn_wmma_f32_16x16x32_bf16(
        false, A.v, false, Bm.v, (short)0, acc, false, false);
  }

  // D rows M = r + 8*hi  ->  t = 2*mtile + hi, b = r ; N = lo
  const int tt = mtile * 2 + hi;
  float* dst = zfrag +
      (((((size_t)dir * Tlen + tt) * 16 + ntile) * 16 + lo) * 8);
  *(v8f*)dst = acc;
}

// ---------------------------------------------------------------------------
// Kernel 2: the serial scans. blockIdx.x = direction, 512 threads = 16 waves,
// wave w owns H columns [16w, 16w+16). h kept double-buffered in LDS (bf16).
// Rows 8..15 of every tile are batch padding: their values never influence
// rows 0..7 (row m of D depends only on row m of A), so no masking needed.
// ---------------------------------------------------------------------------
__global__ void __launch_bounds__(512)
rnn_scan(const float* __restrict__ zfrag, const uint16_t* __restrict__ wt,
         float* __restrict__ out) {
  __shared__ uint16_t hbuf[2][16 * HROW];

  const int dir  = blockIdx.x;
  const int wave = threadIdx.x >> 5;
  const int lane = threadIdx.x & 31;
  const int lo = lane & 15, hi = lane >> 4;
  const int ntile = wave;

  for (int i = threadIdx.x; i < 2 * 16 * HROW; i += 512)
    hbuf[0][i] = 0;                      // zero both buffers (h0 = 0)

  // Preload all 8 B-fragments of W_hh(dir) for this wave's N-tile: 64 VGPRs,
  // reused across all 4096 steps.
  const uint16_t* brow =
      wt + (dir ? 3 * 65536 : 1 * 65536) + (size_t)(ntile * 16 + lo) * 256;
  Frag16 Bf[8];
#pragma unroll
  for (int kk = 0; kk < 8; ++kk)
#pragma unroll
    for (int v = 0; v < 8; ++v)
      Bf[kk].u[v] = *(const uint32_t*)(brow + kk * 32 + kpat(v, hi));

  __syncthreads();

  const size_t jcol = (size_t)dir * 256 + ntile * 16 + lo;
  int cur = 0;

  for (int s = 0; s < Tlen; ++s) {
    const int zt = dir ? (Tlen - 1 - s) : s;

    // C <- Z tile (real rows 0..7 live in lanes 0..15); pad rows get zero.
    v8f acc;
    if (hi == 0) {
      const float* zsrc = zfrag +
          (((((size_t)dir * Tlen + zt) * 16 + ntile) * 16 + lo) * 8);
      acc = *(const v8f*)zsrc;
      // prefetch next step's Z fragment while we grind WMMAs
      const int zn = dir ? (zt - 1) : (zt + 1);
      if (zn >= 0 && zn < Tlen)
        __builtin_prefetch(zfrag +
            (((((size_t)dir * Tlen + zn) * 16 + ntile) * 16 + lo) * 8), 0, 0);
    } else {
#pragma unroll
      for (int r = 0; r < 8; ++r) acc[r] = 0.0f;
    }

    // acc += h_{s-1} @ W_hh : 8 chained bf16 WMMAs over K=256
#pragma unroll
    for (int kk = 0; kk < 8; ++kk) {
      Frag16 A;
      const uint16_t* hrow = &hbuf[cur][lo * HROW + kk * 32 + hi * 8];
#pragma unroll
      for (int v = 0; v < 8; ++v) {
        const int kq = (v < 4) ? (v * 2) : (16 + (v - 4) * 2);
        A.u[v] = *(const uint32_t*)(hrow + kq);
      }
      acc = __builtin_amdgcn_wmma_f32_16x16x32_bf16(
          false, A.v, false, Bf[kk].v, (short)0, acc, false, false);
    }

    // Write h_s (bf16) into the other LDS buffer. D layout: VGPR r, this lane
    // -> element (M = r + 8*hi, N = lo), column c = ntile*16 + lo.
    const int nxt = cur ^ 1;
    uint16_t* hw = &hbuf[nxt][(hi * 8) * HROW + ntile * 16 + lo];
#pragma unroll
    for (int r = 0; r < 8; ++r)
      hw[r * HROW] = f32_to_bf16_rne(acc[r]);

    // Emit output rows 0..7 (f32): out[b][t][dir*256 + j]
    if (hi == 0) {
#pragma unroll
      for (int r = 0; r < 8; ++r)
        out[((size_t)r * Tlen + s) * 512 + jcol] = acc[r];
    }

    __syncthreads();
    cur = nxt;
  }
}

// ---------------------------------------------------------------------------
// Host-side launcher
// ---------------------------------------------------------------------------
extern "C" void kernel_launch(void* const* d_in, const int* in_sizes, int n_in,
                              void* d_out, int out_size, void* d_ws, size_t ws_size,
                              hipStream_t stream) {
  const int*   X     = (const int*)d_in[0];
  const float* emb   = (const float*)d_in[1];
  const float* W_hx  = (const float*)d_in[2];
  const float* W_hh  = (const float*)d_in[3];
  const float* b_h   = (const float*)d_in[4];
  const float* W_hx_ = (const float*)d_in[5];
  const float* W_hh_ = (const float*)d_in[6];
  const float* b_h_  = (const float*)d_in[7];
  float* out = (float*)d_out;

  uint16_t* wt    = (uint16_t*)d_ws;                             // 512 KB
  uint16_t* embb  = (uint16_t*)((char*)d_ws + WT_BYTES);         // 16 MB
  float*    zfrag = (float*)((char*)d_ws + WT_BYTES + EMB_BYTES);// 64 MB

  // 0a) bf16 weight transposes (fragment-friendly [n][k] layout)
  rnn_wconv<<<(int)(WT_ELEMS / 256), 256, 0, stream>>>(W_hx, W_hh, W_hx_, W_hh_, wt);

  // 0b) bf16 embedding copy (pure bandwidth; removes cvt VALU from the GEMM)
  {
    int pairs = (int)(EMB_ELEMS / 2);
    rnn_econv<<<(pairs + 255) / 256, 256, 0, stream>>>(emb, (uint32_t*)embb);
  }

  // 1) Z = gather(emb, X) @ W_hx(dir) + b(dir), both directions
  //    65536 tiles -> 8192 blocks x 8 waves
  rnn_zx_gemm<<<8192, 256, 0, stream>>>(X, embb, b_h, b_h_, wt, zfrag);

  // 2) the two serial scans (one workgroup each)
  rnn_scan<<<2, 512, 0, stream>>>(zfrag, wt, out);
}
